// SRU_78898549227988
// MI455X (gfx1250) — compile-verified, matrix operand order checked by reading, not run on hardware
//
#include <hip/hip_runtime.h>
#include <hip/hip_bf16.h>
#include <cmath>

// ---------------------------------------------------------------------------
// SRU on MI455X (gfx1250, wave32).
//   Phase 0: Wt[n][k] = f16(W[k][n])  (6 MB, LDS-tiled transpose, ~1 us)
//   Phase 1: U[M=32768, N=3072] = X * W  via v_wmma_f32_16x16x32_f16,
//            128x128 block tile, BK=32, register double-buffered.
//   Phase 2: bidirectional elementwise scan over T (memory/latency bound).
// ---------------------------------------------------------------------------

typedef _Float16 v16h __attribute__((ext_vector_type(16)));
typedef _Float16 v8h  __attribute__((ext_vector_type(8)));
typedef float    v8f  __attribute__((ext_vector_type(8)));

#define B_  16
#define T_  2048
#define D_  1024
#define H_  512
#define M_  (B_ * T_)        // 32768 rows (m = b*T + t)
#define N_  (2 * H_ * 3)     // 3072 cols (n = z*H*3 + h*3 + k)
#define K_  D_               // 1024

#define BM 128
#define BN 128
#define BK 32
#define LDSP (BK + 8)        // 80B row stride: 16B-aligned, staggers banks

// ---------------- Phase 0: W [K_,N_] f32 -> Wt [N_,K_] f16 ----------------
__global__ __launch_bounds__(256)
void sru_wt_kernel(const float* __restrict__ W, _Float16* __restrict__ Wt)
{
    __shared__ float tile[32][33];
    const int nb = blockIdx.x * 32;
    const int kb = blockIdx.y * 32;
    const int tx = threadIdx.x & 31;
    const int ty = threadIdx.x >> 5;      // 0..7
    #pragma unroll
    for (int j = 0; j < 4; ++j)           // coalesced f32 reads along n
        tile[ty + 8 * j][tx] = W[(size_t)(kb + ty + 8 * j) * N_ + nb + tx];
    __syncthreads();
    #pragma unroll
    for (int j = 0; j < 4; ++j)           // coalesced f16 writes along k
        Wt[(size_t)(nb + ty + 8 * j) * K_ + kb + tx] = (_Float16)tile[tx][ty + 8 * j];
}

// ---------------- Phase 1: WMMA GEMM ----------------
__global__ __launch_bounds__(256)
void sru_gemm_kernel(const float*    __restrict__ X,   // [M_, K_] f32
                     const _Float16* __restrict__ Wt,  // [N_, K_] f16 (pre-transposed)
                     float*          __restrict__ U)   // [M_, N_] f32
{
    __shared__ __align__(16) _Float16 As[BM][LDSP];
    __shared__ __align__(16) _Float16 Bs[BN][LDSP];

    const int tid  = threadIdx.x;
    const int lane = tid & 31;
    const int warp = tid >> 5;            // 8 waves (wave32)
    const int lrow = lane & 15;
    const int kh   = lane >> 4;           // half-wave K selector

    const int n0 = blockIdx.x * BN;       // N fastest: concurrent blocks share A via L2
    const int m0 = blockIdx.y * BM;

    const int waveM = warp & 3;           // 4 waves along M: 32 rows each
    const int waveN = warp >> 2;          // 2 waves along N: 64 cols each

    // per-thread staging assignments (fixed across K iterations)
    int aRow[4], aCg[4];
    #pragma unroll
    for (int it = 0; it < 4; ++it) {      // A: 128 rows x 8 float4 groups = 1024
        int idx = it * 256 + tid;
        aRow[it] = idx >> 3;
        aCg[it]  = idx & 7;
    }
    int bRow[2], bCg[2];
    #pragma unroll
    for (int it = 0; it < 2; ++it) {      // B: 128 rows x 4 v8h groups = 512
        int idx = it * 256 + tid;
        bRow[it] = idx >> 2;
        bCg[it]  = idx & 3;
    }

    float4 aReg[4];
    v8h    bReg[2];

    auto loadGlobal = [&](int k0) {
        #pragma unroll
        for (int it = 0; it < 4; ++it)
            aReg[it] = *(const float4*)(X + (size_t)(m0 + aRow[it]) * K_ + k0 + aCg[it] * 4);
        #pragma unroll
        for (int it = 0; it < 2; ++it)
            bReg[it] = *(const v8h*)(Wt + (size_t)(n0 + bRow[it]) * K_ + k0 + bCg[it] * 8);
    };
    auto storeLds = [&]() {
        #pragma unroll
        for (int it = 0; it < 4; ++it) {
            As[aRow[it]][aCg[it] * 4 + 0] = (_Float16)aReg[it].x;
            As[aRow[it]][aCg[it] * 4 + 1] = (_Float16)aReg[it].y;
            As[aRow[it]][aCg[it] * 4 + 2] = (_Float16)aReg[it].z;
            As[aRow[it]][aCg[it] * 4 + 3] = (_Float16)aReg[it].w;
        }
        #pragma unroll
        for (int it = 0; it < 2; ++it)
            *(v8h*)&Bs[bRow[it]][bCg[it] * 8] = bReg[it];
    };

    v8f acc[2][4] = {};

    loadGlobal(0);
    for (int k0 = 0; k0 < K_; k0 += BK) {
        storeLds();
        __syncthreads();

        if (k0 + BK < K_) loadGlobal(k0 + BK);   // overlap HBM with WMMA below

        // 16-bit A/B fragment layout: lane = M/N % 16,
        // K = kh*8 + {0..7} (VGPR0-3) and kh*8 + {16..23} (VGPR4-7)
        v16h afrag[2], bfrag[4];
        #pragma unroll
        for (int mt = 0; mt < 2; ++mt) {
            int row = waveM * 32 + mt * 16 + lrow;
            v8h lo = *(const v8h*)&As[row][kh * 8];
            v8h hi = *(const v8h*)&As[row][kh * 8 + 16];
            #pragma unroll
            for (int i = 0; i < 8; ++i) { afrag[mt][i] = lo[i]; afrag[mt][i + 8] = hi[i]; }
        }
        #pragma unroll
        for (int nt = 0; nt < 4; ++nt) {
            int row = waveN * 64 + nt * 16 + lrow;
            v8h lo = *(const v8h*)&Bs[row][kh * 8];
            v8h hi = *(const v8h*)&Bs[row][kh * 8 + 16];
            #pragma unroll
            for (int i = 0; i < 8; ++i) { bfrag[nt][i] = lo[i]; bfrag[nt][i + 8] = hi[i]; }
        }

        #pragma unroll
        for (int mt = 0; mt < 2; ++mt)
            #pragma unroll
            for (int nt = 0; nt < 4; ++nt)
                acc[mt][nt] = __builtin_amdgcn_wmma_f32_16x16x32_f16(
                    false, afrag[mt], false, bfrag[nt],
                    (short)0, acc[mt][nt], false, false);

        __syncthreads();
    }

    // C/D layout: lane = N % 16, M = kh*8 + vgpr index
    #pragma unroll
    for (int mt = 0; mt < 2; ++mt) {
        #pragma unroll
        for (int nt = 0; nt < 4; ++nt) {
            int col = n0 + waveN * 64 + nt * 16 + lrow;
            #pragma unroll
            for (int r = 0; r < 8; ++r) {
                int row = m0 + waveM * 32 + mt * 16 + kh * 8 + r;
                U[(size_t)row * N_ + col] = acc[mt][nt][r];
            }
        }
    }
}

// ---------------- Phase 2: bidirectional scan ----------------
__device__ __forceinline__ float sigmoidf_(float x) {
    return 1.0f / (1.0f + __expf(-x));
}

__global__ __launch_bounds__(256)
void sru_scan_kernel(const float* __restrict__ U,     // [M_, N_]
                     const float* __restrict__ X,     // [B*T, D]
                     const float* __restrict__ bias,  // [2,2,H]
                     float* __restrict__ out,         // [B, T, 2H]
                     float* __restrict__ c_out)       // [B, 1, 2H]
{
    int gid = blockIdx.x * blockDim.x + threadIdx.x;  // 0 .. 16383
    int h = gid & (H_ - 1);
    int z = (gid >> 9) & 1;
    int b = gid >> 10;
    if (b >= B_) return;

    const float bf = bias[z * 2 * H_ + h];        // bias[z,0,h]
    const float br = bias[z * 2 * H_ + H_ + h];   // bias[z,1,h]

    const int ncol = z * (H_ * 3) + h * 3;        // g,f,r contiguous (12B)
    const int xcol = z * H_ + h;

    const int tstart = z ? (T_ - 1) : 0;
    const int tstep  = z ? -1 : 1;

    // software pipeline: prefetch t's operands one step ahead
    float ug, uf, ur, xv;
    {
        size_t m = (size_t)b * T_ + tstart;
        const float* u = U + m * (size_t)N_ + ncol;
        ug = u[0]; uf = u[1]; ur = u[2];
        xv = X[m * (size_t)D_ + xcol];
    }

    float c = 0.0f;
    for (int i = 0; i < T_; ++i) {
        int t = tstart + i * tstep;
        size_t m = (size_t)b * T_ + t;

        float g = ug, fi = uf, ri = ur, xr = xv;
        if (i + 1 < T_) {                          // issue next loads before the math
            size_t m2 = (size_t)(m + tstep);
            const float* u2 = U + m2 * (size_t)N_ + ncol;
            ug = u2[0]; uf = u2[1]; ur = u2[2];
            xv = X[m2 * (size_t)D_ + xcol];
        }

        float f = sigmoidf_(fi + bf);
        float r = sigmoidf_(ri + br);
        c = f * c + (1.0f - f) * g;
        float hv = r * tanhf(c) + (1.0f - r) * xr;
        out[m * (size_t)(2 * H_) + xcol] = hv;
    }
    c_out[b * (2 * H_) + xcol] = c;
}

extern "C" void kernel_launch(void* const* d_in, const int* in_sizes, int n_in,
                              void* d_out, int out_size, void* d_ws, size_t ws_size,
                              hipStream_t stream) {
    const float* x    = (const float*)d_in[0];  // [B, T, D] == [B*T, D] rows m=b*T+t
    const float* W    = (const float*)d_in[1];  // [D, 2, H, 3] == [D, 3072]
    const float* bias = (const float*)d_in[2];  // [2, 2, H]

    float* out   = (float*)d_out;                              // [B, T, 2H]
    float* c_out = out + (size_t)B_ * T_ * 2 * H_;             // [B, 1, 2H]

    float*     U  = (float*)d_ws;                              // [M_, N_] f32 (384 MB)
    _Float16*  Wt = (_Float16*)((char*)d_ws + (size_t)M_ * N_ * sizeof(float)); // 6 MB

    dim3 tgrid(N_ / 32, K_ / 32);  // 96 x 32
    sru_wt_kernel<<<tgrid, 256, 0, stream>>>(W, Wt);

    dim3 ggrid(N_ / BN, M_ / BM);  // 24 x 256, N fastest for L2 reuse of A
    sru_gemm_kernel<<<ggrid, 256, 0, stream>>>(x, Wt, U);

    sru_scan_kernel<<<(B_ * 2 * H_) / 256, 256, 0, stream>>>(U, x, bias, out, c_out);
}
